// XdiffPerciever_60404420051703
// MI455X (gfx1250) — matrix operand order repaired and checked by reading, not compiled, not run on hardware
//
#include <hip/hip_runtime.h>
#include <math.h>

// ---------------- model dimensions (fixed by the reference) ----------------
#define BB    4      // batch
#define NTOK  2048   // N input tokens
#define LQ    256    // L latents
#define DM    512    // d_model == d_cross
#define NH    8      // heads
#define DHEAD 64     // DM / NH
#define DFFC  2048   // dff (also dff_final)
#define NXFC  40     // 4*K_XDIFF
#define XF    320    // NH * NXFC
#define NOUT  4

// d_in layout (setup_inputs() dict order, params flattened in insertion order):
// 0 inputs, 1 x_inputs, 2 x_latents, 3 x_token,
// 4..21 cross params, 22..39 self params,
// 40 gf, 41 bf, 42 wf0, 43 cf0, 44 wf1, 45 cf1
enum { PW_Q = 0, PB_Q, PW_K, PB_K, PW_V, PB_V, PW_X, PB_X, PW_O, PB_O,
       PW_0, PC_0, PW_1, PC_1, PG_1, PBE_1, PG_2, PBE_2 };

// GEMM epilogue flags
#define GF_GELU  1   // exact GELU
#define GF_C32   2   // C is f32 (else f16)
#define GF_ADD16 4   // += f16 Add[m,n]
#define GF_ADD32 8   // += f32 Add[m,n]
#define GF_CT    16  // store transposed: C[n*ldc + m] as f16

typedef _Float16 v16h __attribute__((ext_vector_type(16)));
typedef _Float16 v8h  __attribute__((ext_vector_type(8)));
typedef float    v8f  __attribute__((ext_vector_type(8)));

union U16 { v16h v; v8h h[2]; };

__device__ __forceinline__ float gelu_exact(float x) {
  return 0.5f * x * (1.0f + erff(x * 0.70710678118654752f));
}

__device__ __forceinline__ v8f wmma16x16x32(v16h a, v16h b, v8f c) {
  return __builtin_amdgcn_wmma_f32_16x16x32_f16(false, a, false, b, (short)0, c,
                                                false, false);
}

// Async-DMA one 16-byte chunk of a 64x32-half tile (row-major, ld halfs) into
// LDS laid out [row][40] halfs. chunk c in [0,256): row = c>>2, off = (c&3)*8.
__device__ __forceinline__ void async_chunk(const _Float16* g, int ld,
                                            _Float16* lds, int c) {
  const int row = c >> 2;
  const int off = (c & 3) << 3;                    // 0,8,16,24 halfs
  const _Float16* src = g + (size_t)row * ld + off;
  const unsigned dst = (unsigned)(size_t)(lds + row * 40 + off);
  asm volatile("global_load_async_to_lds_b128 %0, %1, off"
               :: "v"(dst), "v"(src) : "memory");
}

// One full 64x32-half tile: 256 chunks over 128 threads -> 2 unconditional
// issues per thread (no exec masking, no branches).
__device__ __forceinline__ void async_tile(const _Float16* g, int ld,
                                           _Float16* lds, int tid) {
  async_chunk(g, ld, lds, tid);
  async_chunk(g, ld, lds, tid + 128);
}

// ---------------------------------------------------------------------------
// Batched f16 GEMM: C = epi(alpha * A(MxK) * B^T + bias[n] + Add[m,n])
// A: (M x K) f16 row-major; B: (N x K) f16 row-major (pre-transposed weights /
// K-contiguous activations). M%64==0, N%64==0, K%32==0.
// 128 threads (4 waves), 64x64 tile, K-step 32, double-buffered async-LDS DMA.
// Batched via blockIdx.z: offset = (z/innerDiv)*s?o + (z%innerDiv)*s?i
// ---------------------------------------------------------------------------
__global__ __launch_bounds__(128)
void gemm_wmma_kernel(const _Float16* A, const _Float16* Bm, const float* bias,
                      const void* Add, void* C,
                      int M, int Nc, int K,
                      int lda, int ldb, int ldc, int ldadd,
                      long long sAo, long long sAi, long long sBo, long long sBi,
                      long long sCo, long long sCi, long long sDo, long long sDi,
                      int innerDiv, float alpha, int flags) {
  __builtin_assume(threadIdx.x < 128u);
  const int z  = blockIdx.z;
  const int zo = z / innerDiv, zi = z - zo * innerDiv;
  A  += zo * sAo + zi * sAi;
  Bm += zo * sBo + zi * sBi;

  const int m0 = blockIdx.x * 64;
  const int n0 = blockIdx.y * 64;

  __shared__ __align__(16) _Float16 As[2][64 * 40];
  __shared__ __align__(16) _Float16 Bs[2][64 * 40];

  const int tid  = threadIdx.x;
  const int lane = tid & 31;
  const int wave = tid >> 5;
  const int wm   = (wave >> 1) * 32;   // wave row offset in tile
  const int wn   = (wave & 1) * 32;    // wave col offset in tile
  const int lr   = lane & 15;
  const int kh   = lane >> 4;          // K-half select per WMMA layout

  v8f acc[2][2];
  for (int i = 0; i < 2; ++i)
    for (int j = 0; j < 2; ++j)
      for (int r = 0; r < 8; ++r) acc[i][j][r] = 0.0f;

  const _Float16* Ab = A + (size_t)m0 * lda;
  const _Float16* Bb = Bm + (size_t)n0 * ldb;

  // prologue: prefetch first K-tile into buffer 0
  async_tile(Ab, lda, As[0], tid);
  async_tile(Bb, ldb, Bs[0], tid);

  int cur = 0;
  for (int k0 = 0; k0 < K; k0 += 32) {
    if ((k0 + 32) < K) {  // prefetch next tile into the other buffer
      async_tile(Ab + (k0 + 32), lda, As[cur ^ 1], tid);
      async_tile(Bb + (k0 + 32), ldb, Bs[cur ^ 1], tid);
      asm volatile("s_wait_asynccnt 0x4" ::: "memory");
    } else {
      asm volatile("s_wait_asynccnt 0x0" ::: "memory");
    }
    __syncthreads();   // current buffer fully resident for all waves

    // fragments per ISA 16-bit A (16x32) / B (32x16) lane layouts
    U16 a[2], b[2];
    for (int i = 0; i < 2; ++i) {
      const _Float16* p = &As[cur][(wm + i * 16 + lr) * 40];
      a[i].h[0] = *(const v8h*)(p + kh * 8);        // K = kh*8 .. +7
      a[i].h[1] = *(const v8h*)(p + 16 + kh * 8);   // K = 16+kh*8 .. +7
    }
    for (int j = 0; j < 2; ++j) {
      const _Float16* p = &Bs[cur][(wn + j * 16 + lr) * 40];
      b[j].h[0] = *(const v8h*)(p + kh * 8);        // K = kh*8 .. +7
      b[j].h[1] = *(const v8h*)(p + 16 + kh * 8);   // K = 16+kh*8 .. +7
    }
    for (int i = 0; i < 2; ++i)
      for (int j = 0; j < 2; ++j)
        acc[i][j] = wmma16x16x32(a[i].v, b[j].v, acc[i][j]);
    __syncthreads();   // all waves done reading before buffer is overwritten
    cur ^= 1;
  }

  // epilogue: C/D layout -> M = r + kh*8, N = lane&15 within each 16x16 tile
  const _Float16* Add16 = (const _Float16*)Add;
  const float*    Add32 = (const float*)Add;
  if (Add) { Add16 += zo * sDo + zi * sDi; Add32 += zo * sDo + zi * sDi; }
  _Float16* C16 = (_Float16*)C;
  float*    C32 = (float*)C;
  C16 += zo * sCo + zi * sCi;
  C32 += zo * sCo + zi * sCi;

  for (int i = 0; i < 2; ++i) {
    const int mb = m0 + wm + i * 16 + kh * 8;
    for (int j = 0; j < 2; ++j) {
      const int n = n0 + wn + j * 16 + lr;
      const float bi = bias ? bias[n] : 0.0f;
      for (int r = 0; r < 8; ++r) {
        const int m = mb + r;
        float v = alpha * acc[i][j][r] + bi;
        if (flags & GF_ADD16) v += (float)Add16[(size_t)m * ldadd + n];
        if (flags & GF_ADD32) v += Add32[(size_t)m * ldadd + n];
        if (flags & GF_GELU)  v = gelu_exact(v);
        if (flags & GF_CT)       C16[(size_t)n * ldc + m] = (_Float16)v;
        else if (flags & GF_C32) C32[(size_t)m * ldc + n] = v;
        else                     C16[(size_t)m * ldc + n] = (_Float16)v;
      }
    }
  }
}

// --------------- weight transpose + f32->f16 convert: (KxN)->(NxK) ---------
__global__ void wtrans_kernel(const float* w, _Float16* wt, int K, int N) {
  const size_t idx = (size_t)blockIdx.x * blockDim.x + threadIdx.x;
  if (idx >= (size_t)K * N) return;
  const int n = (int)(idx / K);
  const int k = (int)(idx - (size_t)n * K);
  wt[idx] = (_Float16)w[(size_t)k * N + n];
}

// ------------------------- LayerNorm + GELU (rows of 512), f32 -> f16 ------
__global__ __launch_bounds__(128)
void ln_gelu_kernel(const float* x, const float* g, const float* bta,
                    _Float16* out) {
  const int row = blockIdx.x;
  const int tid = threadIdx.x;
  const float* xr = x + (size_t)row * DM;
  float v[4], s = 0.0f, s2 = 0.0f;
  for (int i = 0; i < 4; ++i) {
    const float t = xr[tid + i * 128];
    v[i] = t; s += t; s2 += t * t;
  }
  __shared__ float rs[128], rq[128];
  rs[tid] = s; rq[tid] = s2; __syncthreads();
  for (int o = 64; o > 0; o >>= 1) {
    if (tid < o) { rs[tid] += rs[tid + o]; rq[tid] += rq[tid + o]; }
    __syncthreads();
  }
  const float mean = rs[0] * (1.0f / DM);
  const float var  = rq[0] * (1.0f / DM) - mean * mean;
  const float rstd = rsqrtf(var + 1e-3f);
  _Float16* orow = out + (size_t)row * DM;
  for (int i = 0; i < 4; ++i) {
    const int c = tid + i * 128;
    orow[c] = (_Float16)gelu_exact((v[i] - mean) * rstd * g[c] + bta[c]);
  }
}

// ------------------------- elementwise GELU (f32->f16) / broadcast ---------
__global__ void gelu_kernel(const float* x, _Float16* y, size_t n) {
  const size_t i = (size_t)blockIdx.x * blockDim.x + threadIdx.x;
  if (i < n) y[i] = (_Float16)gelu_exact(x[i]);
}

__global__ void bcast_kernel(const float* tok, float* y, size_t n) {
  const size_t i = (size_t)blockIdx.x * blockDim.x + threadIdx.x;
  if (i < n) y[i] = tok[i & (DM - 1)];
}

// ---------------------------------------------------------------------------
// xdiff attention-bias kernel: writes (f16)
//   bias[b,h,q,k] = (2/sqrt(40)) * sum_f xdiff(d(q,k))_f * xh[b,q,h,f]
// into the logits buffer (QK^T GEMM adds it in its epilogue).
// pe mean == 0 exactly and sum(pe^2) = 4*sum(damp^2) (distance independent),
// so the std normalization is the constant sqrt(K/sum(damp^2)).
// ---------------------------------------------------------------------------
__global__ __launch_bounds__(256)
void xlogit_kernel(const float* x_lat, const float* x_oth,
                   const _Float16* xproj, _Float16* out, int Nk) {
  const int q = blockIdx.x;
  const int b = blockIdx.y;
  const int tid = threadIdx.x;
  __shared__ float dA[NH][10], dB[NH][10], xq[3];
  if (tid < 80) {
    const int h = tid / 10, i = tid - h * 10;
    const _Float16* xr = xproj + ((size_t)(b * LQ + q)) * XF + h * NXFC;
    dA[h][i] = (float)xr[i]      - (float)xr[20 + i];
    dB[h][i] = (float)xr[10 + i] - (float)xr[30 + i];
  }
  if (tid < 3) xq[tid] = x_lat[((size_t)(b * LQ + q)) * 3 + tid];
  __syncthreads();

  float band[10], kco[10];
  float sum2 = 0.0f;
  for (int i = 0; i < 10; ++i) {
    const float t  = (float)i * (1.0f / 9.0f);
    const float dm = expf(-3.0f * t);
    sum2 += dm * dm;
  }
  // sqrt(2) * sqrt(K/sum(damp^2)) * 2/sqrt(NXF)
  const float coef = 1.41421356237f * sqrtf(10.0f / sum2) * 0.316227766017f;
  for (int i = 0; i < 10; ++i) {
    const float t = (float)i * (1.0f / 9.0f);
    band[i] = 3.14159265358979f * (1.0f + t * t * 199.0f);  // (2-0.01)/0.01
    kco[i]  = coef * expf(-3.0f * t);
  }
  const float q0 = xq[0], q1 = xq[1], q2 = xq[2];

  for (int k = tid; k < Nk; k += 256) {
    const float* xo = x_oth + ((size_t)b * Nk + k) * 3;
    const float dx = q0 - xo[0], dy = q1 - xo[1], dz = q2 - xo[2];
    const float v = sqrtf(dx * dx + dy * dy + dz * dz) * 0.5f;  // /SCALE_XDIFF
    float acc[NH];
    for (int h = 0; h < NH; ++h) acc[h] = 0.0f;
#pragma unroll
    for (int i = 0; i < 10; ++i) {
      float s, c;
      sincosf(v * band[i], &s, &c);
      const float kc = -c * kco[i];
      const float ks =  s * kco[i];
#pragma unroll
      for (int h = 0; h < NH; ++h) acc[h] += kc * dA[h][i] + ks * dB[h][i];
    }
    for (int h = 0; h < NH; ++h)
      out[(((size_t)b * NH + h) * LQ + q) * (size_t)Nk + k] = (_Float16)acc[h];
  }
}

// ------------------------- row softmax (f16, in place) ---------------------
__global__ __launch_bounds__(256)
void softmax_kernel(_Float16* logits, int Nk) {
  _Float16* p = logits + (size_t)blockIdx.x * Nk;
  const int tid = threadIdx.x;
  __shared__ float red[256];
  float mx = -3.4e38f;
  for (int k = tid; k < Nk; k += 256) mx = fmaxf(mx, (float)p[k]);
  red[tid] = mx; __syncthreads();
  for (int o = 128; o > 0; o >>= 1) {
    if (tid < o) red[tid] = fmaxf(red[tid], red[tid + o]);
    __syncthreads();
  }
  const float m = red[0];
  __syncthreads();
  float s = 0.0f;
  float ev[8];                      // Nk/256 <= 8
  int cnt = 0;
  for (int k = tid; k < Nk; k += 256) {
    const float e = expf((float)p[k] - m);
    ev[cnt++] = e; s += e;
  }
  red[tid] = s; __syncthreads();
  for (int o = 128; o > 0; o >>= 1) {
    if (tid < o) red[tid] += red[tid + o];
    __syncthreads();
  }
  const float inv = 1.0f / red[0];
  cnt = 0;
  for (int k = tid; k < Nk; k += 256) p[k] = (_Float16)(ev[cnt++] * inv);
}

// ------------------------- final head: 2048 -> 4 ---------------------------
__global__ __launch_bounds__(256)
void head_kernel(const _Float16* h, const float* w, const float* c, float* out) {
  const int row = blockIdx.x;
  const int tid = threadIdx.x;
  const _Float16* hr = h + (size_t)row * DFFC;
  float a0 = 0.f, a1 = 0.f, a2 = 0.f, a3 = 0.f;
  for (int k = tid; k < DFFC; k += 256) {
    const float hv = (float)hr[k];
    const float* wr = w + (size_t)k * NOUT;
    a0 += hv * wr[0]; a1 += hv * wr[1]; a2 += hv * wr[2]; a3 += hv * wr[3];
  }
  __shared__ float4 red[256];
  red[tid] = make_float4(a0, a1, a2, a3);
  __syncthreads();
  for (int o = 128; o > 0; o >>= 1) {
    if (tid < o) {
      red[tid].x += red[tid + o].x; red[tid].y += red[tid + o].y;
      red[tid].z += red[tid + o].z; red[tid].w += red[tid + o].w;
    }
    __syncthreads();
  }
  if (tid == 0) {
    out[row * NOUT + 0] = red[0].x + c[0];
    out[row * NOUT + 1] = red[0].y + c[1];
    out[row * NOUT + 2] = red[0].z + c[2];
    out[row * NOUT + 3] = red[0].w + c[3];
  }
}

// ===========================================================================
// host orchestration
// ===========================================================================
struct LayerW {
  const _Float16 *wq, *wk, *wv, *wx, *wo, *w0, *w1;
};

static inline void launch_gemm(hipStream_t st,
    const _Float16* A, const _Float16* Bm, const float* bias, const void* Add,
    void* C, int M, int Nc, int K, int lda, int ldb, int ldc, int ldadd,
    long long sAo, long long sAi, long long sBo, long long sBi,
    long long sCo, long long sCi, long long sDo, long long sDi,
    int innerDiv, int batches, float alpha, int flags) {
  dim3 grid(M / 64, Nc / 64, batches);
  gemm_wmma_kernel<<<grid, dim3(128), 0, st>>>(A, Bm, bias, Add, C,
      M, Nc, K, lda, ldb, ldc, ldadd,
      sAo, sAi, sBo, sBi, sCo, sCi, sDo, sDi, innerDiv, alpha, flags);
}

static void run_layer(hipStream_t st, void* const* d_in, int pb, bool cross,
                      const float* x_lat, const float* x_oth, const LayerW& W,
                      float* latents, _Float16* latbuf, const _Float16* kvg,
                      _Float16* qb, _Float16* kb, _Float16* vbt, _Float16* xb,
                      _Float16* logits, _Float16* attno, _Float16* ffnb) {
  const int Nk  = cross ? NTOK : LQ;
  const int TOT = BB * Nk;
  const float* bq  = (const float*)d_in[pb + PB_Q];
  const float* bk  = (const float*)d_in[pb + PB_K];
  const float* bv  = (const float*)d_in[pb + PB_V];
  const float* bx  = (const float*)d_in[pb + PB_X];
  const float* bo  = (const float*)d_in[pb + PB_O];
  const float* c0  = (const float*)d_in[pb + PC_0];
  const float* c1  = (const float*)d_in[pb + PC_1];
  const float* g1  = (const float*)d_in[pb + PG_1];
  const float* be1 = (const float*)d_in[pb + PBE_1];
  const float* g2  = (const float*)d_in[pb + PG_2];
  const float* be2 = (const float*)d_in[pb + PBE_2];

  // lat = gelu(layernorm(latents)) -> f16
  ln_gelu_kernel<<<BB * LQ, 128, 0, st>>>(latents, g1, be1, latbuf);

  // projections (B = pre-transposed f16 weights, N x K)
  launch_gemm(st, latbuf, W.wq, bq, nullptr, qb, BB * LQ, DM, DM, DM, DM, DM, 0,
              0, 0, 0, 0, 0, 0, 0, 0, 1, 1, 1.0f, 0);
  launch_gemm(st, kvg, W.wk, bk, nullptr, kb, BB * Nk, DM, DM, DM, DM, DM, 0,
              0, 0, 0, 0, 0, 0, 0, 0, 1, 1, 1.0f, 0);
  // V stored transposed: vbt[d][b*Nk + tok], leading dim TOT
  launch_gemm(st, kvg, W.wv, bv, nullptr, vbt, BB * Nk, DM, DM, DM, DM, TOT, 0,
              0, 0, 0, 0, 0, 0, 0, 0, 1, 1, 1.0f, GF_CT);
  launch_gemm(st, qb, W.wx, bx, nullptr, xb, BB * LQ, XF, DM, DM, DM, XF, 0,
              0, 0, 0, 0, 0, 0, 0, 0, 1, 1, 1.0f, 0);

  // xdiff bias -> logits buffer (f16)
  xlogit_kernel<<<dim3(LQ, BB), 256, 0, st>>>(x_lat, x_oth, xb, logits, Nk);

  // logits = QK^T/sqrt(64) + xdiff-bias  (batched over (b,h); in-place add)
  launch_gemm(st, qb, kb, nullptr, logits, logits,
              LQ, Nk, DHEAD, DM, DM, Nk, Nk,
              (long long)LQ * DM, 64,
              (long long)Nk * DM, 64,
              (long long)NH * LQ * Nk, (long long)LQ * Nk,
              (long long)NH * LQ * Nk, (long long)LQ * Nk,
              NH, BB * NH, 0.125f, GF_ADD16);

  softmax_kernel<<<BB * NH * LQ, 256, 0, st>>>(logits, Nk);

  // attn @ V -> attno (B, L, H*64): B operand rows = head dims of vbt
  launch_gemm(st, logits, vbt, nullptr, nullptr, attno,
              LQ, DHEAD, Nk, Nk, TOT, DM, 0,
              (long long)NH * LQ * Nk, (long long)LQ * Nk,
              Nk, (long long)64 * TOT,
              (long long)LQ * DM, 64, 0, 0,
              NH, BB * NH, 1.0f, 0);

  // latents = latents + attno @ wo + bo   (f32 residual, in place)
  launch_gemm(st, attno, W.wo, bo, latents, latents, BB * LQ, DM, DM, DM, DM, DM,
              DM, 0, 0, 0, 0, 0, 0, 0, 0, 1, 1, 1.0f, GF_ADD32 | GF_C32);

  // FFN with pre-LN+GELU and f32 residual
  ln_gelu_kernel<<<BB * LQ, 128, 0, st>>>(latents, g2, be2, latbuf);
  launch_gemm(st, latbuf, W.w0, c0, nullptr, ffnb, BB * LQ, DFFC, DM, DM, DM,
              DFFC, 0, 0, 0, 0, 0, 0, 0, 0, 0, 1, 1, 1.0f, GF_GELU);
  launch_gemm(st, ffnb, W.w1, c1, latents, latents, BB * LQ, DM, DFFC, DFFC,
              DFFC, DM, DM, 0, 0, 0, 0, 0, 0, 0, 0, 1, 1, 1.0f,
              GF_ADD32 | GF_C32);
}

extern "C" void kernel_launch(void* const* d_in, const int* in_sizes, int n_in,
                              void* d_out, int out_size, void* d_ws, size_t ws_size,
                              hipStream_t stream) {
  (void)in_sizes; (void)n_in; (void)out_size; (void)ws_size;

  const float* inputs    = (const float*)d_in[0];
  const float* x_inputs  = (const float*)d_in[1];
  const float* x_latents = (const float*)d_in[2];
  const float* x_token   = (const float*)d_in[3];
  const int CROSS = 4, SELF = 22;
  const float* gf  = (const float*)d_in[40];
  const float* bf  = (const float*)d_in[41];
  const float* wf0 = (const float*)d_in[42];
  const float* cf0 = (const float*)d_in[43];
  const float* wf1 = (const float*)d_in[44];
  const float* cf1 = (const float*)d_in[45];

  // ---- workspace carve-up (all chunks 16B aligned) ----
  char* p = (char*)d_ws;
  auto aF32 = [&](size_t n) { float* r = (float*)p; p += n * sizeof(float); return r; };
  auto aF16 = [&](size_t n) { _Float16* r = (_Float16*)p; p += n * sizeof(_Float16); return r; };

  float*    latents = aF32((size_t)BB * LQ * DM);
  _Float16* latbuf  = aF16((size_t)BB * LQ * DM);
  _Float16* kvg     = aF16((size_t)BB * NTOK * DM);
  _Float16* qb      = aF16((size_t)BB * LQ * DM);
  _Float16* kb      = aF16((size_t)BB * NTOK * DM);
  _Float16* vbt     = aF16((size_t)DM * BB * NTOK);
  _Float16* xb      = aF16((size_t)BB * LQ * XF);
  _Float16* logits  = aF16((size_t)BB * NH * LQ * NTOK);
  _Float16* attno   = aF16((size_t)BB * LQ * DM);
  _Float16* ffnb    = aF16((size_t)BB * LQ * DFFC);

  // ---- transpose+convert all GEMM weights to f16 (N x K) ----
  auto wt = [&](int idx, int K, int N) {
    _Float16* dst = aF16((size_t)K * N);
    const size_t tot = (size_t)K * N;
    wtrans_kernel<<<(unsigned)((tot + 255) / 256), 256, 0, stream>>>(
        (const float*)d_in[idx], dst, K, N);
    return (const _Float16*)dst;
  };
  LayerW LW[2];
  for (int li = 0; li < 2; ++li) {
    const int pb = li == 0 ? CROSS : SELF;
    LW[li].wq = wt(pb + PW_Q, DM, DM);
    LW[li].wk = wt(pb + PW_K, DM, DM);
    LW[li].wv = wt(pb + PW_V, DM, DM);
    LW[li].wx = wt(pb + PW_X, DM, XF);
    LW[li].wo = wt(pb + PW_O, DM, DM);
    LW[li].w0 = wt(pb + PW_0, DM, DFFC);
    LW[li].w1 = wt(pb + PW_1, DFFC, DM);
  }
  const _Float16* wf0t = [&] {
    _Float16* dst = aF16((size_t)DM * DFFC);
    wtrans_kernel<<<(unsigned)(((size_t)DM * DFFC + 255) / 256), 256, 0, stream>>>(
        wf0, dst, DM, DFFC);
    return (const _Float16*)dst;
  }();

  constexpr size_t SZ_LAT = (size_t)BB * LQ * DM;
  constexpr size_t SZ_KV  = (size_t)BB * NTOK * DM;
  // kv for cross layers: gelu(raw inputs) -> f16, reused across both repeats
  gelu_kernel<<<(unsigned)((SZ_KV + 255) / 256), 256, 0, stream>>>(inputs, kvg, SZ_KV);
  // latents init: broadcast x_token (f32 residual stream)
  bcast_kernel<<<(unsigned)((SZ_LAT + 255) / 256), 256, 0, stream>>>(x_token, latents, SZ_LAT);

  for (int rep = 0; rep < 2; ++rep) {
    run_layer(stream, d_in, CROSS, true, x_latents, x_inputs, LW[0],
              latents, latbuf, kvg, qb, kb, vbt, xb, logits, attno, ffnb);
    for (int sl = 0; sl < 2; ++sl)
      run_layer(stream, d_in, SELF, false, x_latents, x_latents, LW[1],
                latents, latbuf, latbuf, qb, kb, vbt, xb, logits, attno, ffnb);
  }

  // final head
  ln_gelu_kernel<<<BB * LQ, 128, 0, stream>>>(latents, gf, bf, latbuf);
  launch_gemm(stream, latbuf, wf0t, cf0, nullptr, ffnb, BB * LQ, DFFC, DM, DM,
              DM, DFFC, 0, 0, 0, 0, 0, 0, 0, 0, 0, 1, 1, 1.0f, GF_GELU);
  head_kernel<<<BB * LQ, 256, 0, stream>>>(ffnb, wf1, cf1, (float*)d_out);
}
